// VoxelToPointTransformerLayer_10342281248858
// MI455X (gfx1250) — matrix operand order repaired and checked by reading, not compiled, not run on hardware
//
#include <hip/hip_runtime.h>
#include <math.h>

typedef __bf16 bf16;
typedef __attribute__((ext_vector_type(16))) __bf16 v16bf;
typedef __attribute__((ext_vector_type(8)))  __bf16 v8bf;
typedef __attribute__((ext_vector_type(8)))  float  v8f;

#define LN_EPS 1e-5f

// LDS row strides with bank-conflict padding
#define AS_STRIDE 264   // bf16 halves per A row (256 + 8, keeps 16B alignment)
#define CS_STRIDE 260   // f32 per row (256 + 4)

// ---------------------------------------------------------------------------
// f32 -> bf16 conversion (row-major, for A-side activations)
// ---------------------------------------------------------------------------
__global__ void cvt_f32_bf16_kernel(const float4* __restrict__ in,
                                    bf16* __restrict__ out, int n4) {
  int i = blockIdx.x * blockDim.x + threadIdx.x;
  if (i < n4) {
    float4 f = in[i];
    int o = i * 4;
    out[o + 0] = (bf16)f.x;
    out[o + 1] = (bf16)f.y;
    out[o + 2] = (bf16)f.z;
    out[o + 3] = (bf16)f.w;
  }
}

// ---------------------------------------------------------------------------
// Pack a row-major f32 weight [K x N] into bf16 WMMA-B fragment order:
//   PB[((ktile*NT + ntile)*32 + lane)*16 + t]
//     = W[(ktile*32 + (t&8)*2 + (lane>>4)*8 + (t&7)) * N + ntile*16 + (lane&15)]
// so a wave's B fragment is one contiguous 1KB block; per lane one 32B load.
// ---------------------------------------------------------------------------
__global__ void pack_b_kernel(const float* __restrict__ W, bf16* __restrict__ PB,
                              int K, int N) {
  int i = blockIdx.x * blockDim.x + threadIdx.x;
  if (i >= K * N) return;
  int t     = i & 15;
  int l     = (i >> 4) & 31;
  int rem   = i >> 9;
  int NT    = N >> 4;
  int ntile = rem % NT;
  int ktile = rem / NT;
  int n  = ntile * 16 + (l & 15);
  int kk = ktile * 32 + (t & 8) * 2 + (l >> 4) * 8 + (t & 7);
  PB[i] = (bf16)W[(size_t)kk * N + n];
}

// A fragment: two contiguous 16B groups per lane (CDNA5 16-bit A 16x32 layout)
__device__ __forceinline__ v16bf load_a_frag(const bf16* row) {
  v8bf lo = *(const v8bf*)(row);
  v8bf hi = *(const v8bf*)(row + 16);
  return __builtin_shufflevector(lo, hi, 0,1,2,3,4,5,6,7,8,9,10,11,12,13,14,15);
}

// ---------------------------------------------------------------------------
// Generic bf16 WMMA GEMM: C[32 x 256-block] per workgroup, 8 waves,
// each wave: 2 M-subtiles x 2 N-tiles = 4 WMMAs per fragment set.
// A = [A0 | A1] split at K0 (multiple of 256). PB = packed B (fragment order).
// mode 0: store f32            (q projection)
// mode 1: LayerNorm -> bf16    (message = LN(out@Wm))   [Ntot==256, grid.y==1]
// mode 2: exact GELU -> bf16   (FFN hidden)
// mode 3: LayerNorm + residual -> f32 (final output)    [Ntot==256, grid.y==1]
// ---------------------------------------------------------------------------
__global__ __launch_bounds__(256)
void gemm16_bf16_kernel(const bf16* __restrict__ A0, const bf16* __restrict__ A1,
                        int K0, int K,
                        const bf16* __restrict__ PB, int Ntot,
                        float* __restrict__ outF, bf16* __restrict__ outB,
                        const float* __restrict__ gamma, const float* __restrict__ beta,
                        const float* __restrict__ resid, int mode) {
  __shared__ __align__(16) bf16 As[32 * AS_STRIDE];  // 16.9 KB A panel (32 x 256)
  __shared__ float Cs[32 * CS_STRIDE];               // 33.3 KB (LN staging)
  __shared__ float mu_s[32], rs_s[32];

  const int tid   = threadIdx.x;
  const int lane  = tid & 31;
  const int wave  = tid >> 5;
  const int mbase = blockIdx.x * 32;
  const int nbase = blockIdx.y * 256;
  const int NT    = Ntot >> 4;                // n-tiles across full N
  const int tile0 = (nbase >> 4) + wave * 2;  // global n-tile ids for this wave
  const int tile1 = tile0 + 1;

  v8f acc[2][2] = {};                         // [m-subtile][n-tile]
  const int nn   = lane & 15;
  const int mrow = (lane < 16) ? 0 : 8;
  const int akb  = (lane < 16) ? 0 : 8;

  const int npan = K >> 8;                    // 256-wide K panels
  for (int pp = 0; pp < npan; ++pp) {
    const int kpan = pp << 8;
    __syncthreads();
    // --- stage A panel 32x256 (each 256-wide panel lies fully in A0 or A1) ---
    {
      const bf16* Asrc; int ldas, rowoff;
      if (kpan < K0) { Asrc = A0; ldas = K0;     rowoff = kpan; }
      else           { Asrc = A1; ldas = K - K0; rowoff = kpan - K0; }
      for (int i = tid; i < 1024; i += 256) {    // 1024 x uint4 = 16 KB
        int m = i >> 5, j = i & 31;
        *(uint4*)(As + m * AS_STRIDE + j * 8) =
            *(const uint4*)(Asrc + (size_t)(mbase + m) * ldas + rowoff + j * 8);
      }
    }
    __syncthreads();

    // prefetch next panel of packed B into L2 (global_prefetch_b8)
    if (pp + 1 < npan) {
      size_t nktile = (size_t)((kpan >> 5) + 8);
      __builtin_prefetch(PB + ((nktile * NT + tile0) * 32 + lane) * 16, 0, 1);
    }

    #pragma unroll
    for (int kk = 0; kk < 8; ++kk) {
      const int ktile = (kpan >> 5) + kk;
      v16bf a0 = load_a_frag(As + nn * AS_STRIDE + kk * 32 + akb);
      v16bf a1 = load_a_frag(As + (16 + nn) * AS_STRIDE + kk * 32 + akb);
      v16bf b0 = *(const v16bf*)(PB + (((size_t)ktile * NT + tile0) * 32 + lane) * 16);
      v16bf b1 = *(const v16bf*)(PB + (((size_t)ktile * NT + tile1) * 32 + lane) * 16);
      acc[0][0] = __builtin_amdgcn_wmma_f32_16x16x32_bf16(false, a0, false, b0, (short)0, acc[0][0], false, false);
      acc[0][1] = __builtin_amdgcn_wmma_f32_16x16x32_bf16(false, a0, false, b1, (short)0, acc[0][1], false, false);
      acc[1][0] = __builtin_amdgcn_wmma_f32_16x16x32_bf16(false, a1, false, b0, (short)0, acc[1][0], false, false);
      acc[1][1] = __builtin_amdgcn_wmma_f32_16x16x32_bf16(false, a1, false, b1, (short)0, acc[1][1], false, false);
    }
  }

  if (mode == 0 || mode == 2) {
    #pragma unroll
    for (int ms = 0; ms < 2; ++ms)
    #pragma unroll
    for (int nt = 0; nt < 2; ++nt)
    #pragma unroll
    for (int i = 0; i < 8; ++i) {
      int mg = mbase + ms * 16 + mrow + i;
      int ng = nbase + (wave * 2 + nt) * 16 + nn;
      float x = acc[ms][nt][i];
      if (mode == 2) {
        x = 0.5f * x * (1.0f + erff(x * 0.70710678118654752f));
        outB[(size_t)mg * Ntot + ng] = (bf16)x;
      } else {
        outF[(size_t)mg * Ntot + ng] = x;
      }
    }
  } else {
    // LayerNorm modes: Ntot == 256, nbase == 0
    #pragma unroll
    for (int ms = 0; ms < 2; ++ms)
    #pragma unroll
    for (int nt = 0; nt < 2; ++nt)
    #pragma unroll
    for (int i = 0; i < 8; ++i)
      Cs[(ms * 16 + mrow + i) * CS_STRIDE + (wave * 2 + nt) * 16 + nn] = acc[ms][nt][i];
    __syncthreads();
    if (tid < 32) {
      float s = 0.f, s2 = 0.f;
      for (int j = 0; j < 256; ++j) { float x = Cs[tid * CS_STRIDE + j]; s += x; s2 += x * x; }
      float mu  = s * (1.0f / 256.0f);
      float var = s2 * (1.0f / 256.0f) - mu * mu;
      mu_s[tid] = mu;
      rs_s[tid] = rsqrtf(var + LN_EPS);
    }
    __syncthreads();
    for (int i = tid; i < 8192; i += 256) {
      int m = i >> 8, n = i & 255;
      float x = (Cs[m * CS_STRIDE + n] - mu_s[m]) * rs_s[m] * gamma[n] + beta[n];
      int mg = mbase + m;
      if (mode == 1) outB[(size_t)mg * 256 + n] = (bf16)x;
      else           outF[(size_t)mg * 256 + n] = resid[(size_t)mg * 256 + n] + x;
    }
  }
}

// ---------------------------------------------------------------------------
// Fused attention: one workgroup per PAIR of voxels (B-fragment reuse: the
// Wk/Wv fragments feed 8 WMMAs per k-tile). K/V LDS reused across the two
// voxels' softmax phases; accumulators for both live in registers.
// ---------------------------------------------------------------------------
__global__ __launch_bounds__(256)
void attn_kernel(const float* __restrict__ target, const float* __restrict__ qF,
                 const float* __restrict__ wts, const float* __restrict__ conf,
                 const bf16* __restrict__ WkP, const bf16* __restrict__ WvP,
                 bf16* __restrict__ aoB) {
  __shared__ __align__(16) bf16 Ts[32 * AS_STRIDE];   // two target tiles (bf16)
  __shared__ float Ks[16 * CS_STRIDE];                // K rows [P][C] (per voxel)
  __shared__ float Vs[16 * CS_STRIDE];                // V rows [P][C] (per voxel)
  __shared__ float qs[2 * 256];
  __shared__ float wsh[32], csh[32];
  __shared__ float coef[8][16];

  const int r0   = blockIdx.x * 2;                    // first voxel of the pair
  const int tid  = threadIdx.x;
  const int lane = tid & 31;
  const int wave = tid >> 5;

  // stage both target tiles (f32 -> bf16), q rows, weights, confidences
  for (int i = tid; i < 2048; i += 256) {
    float4 f = ((const float4*)(target + (size_t)r0 * 4096))[i];
    int o = i * 4, m = o >> 8, c = o & 255;
    bf16* d = Ts + m * AS_STRIDE + c;
    d[0] = (bf16)f.x; d[1] = (bf16)f.y; d[2] = (bf16)f.z; d[3] = (bf16)f.w;
  }
  qs[tid]       = qF[(size_t)r0 * 256 + tid];
  qs[256 + tid] = qF[(size_t)r0 * 256 + 256 + tid];
  if (tid < 32) { wsh[tid] = wts[(size_t)r0 * 16 + tid]; csh[tid] = conf[(size_t)r0 * 16 + tid]; }
  __syncthreads();

  v8f accK[2][2] = {}, accV[2][2] = {};               // [voxel][n-tile]
  const int nn   = lane & 15;
  const int mrow = (lane < 16) ? 0 : 8;
  const int akb  = (lane < 16) ? 0 : 8;
  const int t0   = wave * 2, t1 = wave * 2 + 1;       // NT = 16

  #pragma unroll
  for (int ks = 0; ks < 8; ++ks) {
    v16bf a0 = load_a_frag(Ts + nn * AS_STRIDE + ks * 32 + akb);
    v16bf a1 = load_a_frag(Ts + (16 + nn) * AS_STRIDE + ks * 32 + akb);
    v16bf bk0 = *(const v16bf*)(WkP + (((size_t)ks * 16 + t0) * 32 + lane) * 16);
    v16bf bk1 = *(const v16bf*)(WkP + (((size_t)ks * 16 + t1) * 32 + lane) * 16);
    v16bf bv0 = *(const v16bf*)(WvP + (((size_t)ks * 16 + t0) * 32 + lane) * 16);
    v16bf bv1 = *(const v16bf*)(WvP + (((size_t)ks * 16 + t1) * 32 + lane) * 16);
    accK[0][0] = __builtin_amdgcn_wmma_f32_16x16x32_bf16(false, a0, false, bk0, (short)0, accK[0][0], false, false);
    accK[0][1] = __builtin_amdgcn_wmma_f32_16x16x32_bf16(false, a0, false, bk1, (short)0, accK[0][1], false, false);
    accK[1][0] = __builtin_amdgcn_wmma_f32_16x16x32_bf16(false, a1, false, bk0, (short)0, accK[1][0], false, false);
    accK[1][1] = __builtin_amdgcn_wmma_f32_16x16x32_bf16(false, a1, false, bk1, (short)0, accK[1][1], false, false);
    accV[0][0] = __builtin_amdgcn_wmma_f32_16x16x32_bf16(false, a0, false, bv0, (short)0, accV[0][0], false, false);
    accV[0][1] = __builtin_amdgcn_wmma_f32_16x16x32_bf16(false, a0, false, bv1, (short)0, accV[0][1], false, false);
    accV[1][0] = __builtin_amdgcn_wmma_f32_16x16x32_bf16(false, a1, false, bv0, (short)0, accV[1][0], false, false);
    accV[1][1] = __builtin_amdgcn_wmma_f32_16x16x32_bf16(false, a1, false, bv1, (short)0, accV[1][1], false, false);
  }

  #pragma unroll
  for (int v = 0; v < 2; ++v) {
    __syncthreads();   // previous phase's K/V reads complete before overwrite
    #pragma unroll
    for (int i = 0; i < 8; ++i) {
      Ks[(mrow + i) * CS_STRIDE + t0 * 16 + nn] = accK[v][0][i];
      Ks[(mrow + i) * CS_STRIDE + t1 * 16 + nn] = accK[v][1][i];
      Vs[(mrow + i) * CS_STRIDE + t0 * 16 + nn] = accV[v][0][i];
      Vs[(mrow + i) * CS_STRIDE + t1 * 16 + nn] = accV[v][1][i];
    }
    __syncthreads();

    // scores[h][p] = (q_h . k_ph) * w_p / sqrt(32)
    if (tid < 128) {
      int h = tid >> 4, p = tid & 15;
      float s = 0.f;
      #pragma unroll
      for (int d = 0; d < 32; ++d) s += qs[v * 256 + h * 32 + d] * Ks[p * CS_STRIDE + h * 32 + d];
      coef[h][p] = s * wsh[v * 16 + p] * 0.17677669529663687f;   // 1/sqrt(32)
    }
    __syncthreads();
    // softmax over P, scaled by confidences
    if (tid < 8) {
      int h = tid;
      float mx = -1e30f;
      #pragma unroll
      for (int p = 0; p < 16; ++p) mx = fmaxf(mx, coef[h][p]);
      float e[16]; float sum = 0.f;
      #pragma unroll
      for (int p = 0; p < 16; ++p) { e[p] = expf(coef[h][p] - mx); sum += e[p]; }
      float inv = 1.0f / sum;
      #pragma unroll
      for (int p = 0; p < 16; ++p) coef[h][p] = e[p] * inv * csh[v * 16 + p];
    }
    __syncthreads();
    // out[c] = sum_p coef[h(c)][p] * V[p][c]
    {
      int c = tid, h = c >> 5;
      float o = 0.f;
      #pragma unroll
      for (int p = 0; p < 16; ++p) o += coef[h][p] * Vs[p * CS_STRIDE + c];
      aoB[(size_t)(r0 + v) * 256 + c] = (bf16)o;
    }
  }
}

// ---------------------------------------------------------------------------
extern "C" void kernel_launch(void* const* d_in, const int* in_sizes, int n_in,
                              void* d_out, int out_size, void* d_ws, size_t ws_size,
                              hipStream_t stream) {
  const float* source = (const float*)d_in[0];
  const float* target = (const float*)d_in[1];
  const float* wts    = (const float*)d_in[2];
  const float* confid = (const float*)d_in[3];
  const float* Wq  = (const float*)d_in[4];
  const float* Wk  = (const float*)d_in[5];
  const float* Wv  = (const float*)d_in[6];
  const float* Wm  = (const float*)d_in[7];
  const float* g1  = (const float*)d_in[8];
  const float* b1  = (const float*)d_in[9];
  const float* Wf1 = (const float*)d_in[10];
  const float* Wf2 = (const float*)d_in[11];
  const float* g2  = (const float*)d_in[12];
  const float* b2  = (const float*)d_in[13];
  (void)in_sizes; (void)n_in; (void)out_size; (void)ws_size;

  const int BV = 16384;  // B*V rows

  char* ws = (char*)d_ws;
  size_t off = 0;
  auto carve = [&](size_t bytes) -> void* {
    void* p = ws + off;
    off += (bytes + 255) & ~(size_t)255;
    return p;
  };
  bf16*  WqP  = (bf16*) carve((size_t)256 * 256 * 2);
  bf16*  WkP  = (bf16*) carve((size_t)256 * 256 * 2);
  bf16*  WvP  = (bf16*) carve((size_t)256 * 256 * 2);
  bf16*  WmP  = (bf16*) carve((size_t)256 * 256 * 2);
  bf16*  Wf1P = (bf16*) carve((size_t)512 * 2048 * 2);
  bf16*  Wf2P = (bf16*) carve((size_t)2048 * 256 * 2);
  bf16*  srcB = (bf16*) carve((size_t)BV * 256 * 2);
  float* qF   = (float*)carve((size_t)BV * 256 * 4);
  bf16*  aoB  = (bf16*) carve((size_t)BV * 256 * 2);
  bf16*  m1B  = (bf16*) carve((size_t)BV * 256 * 2);
  bf16*  hB   = (bf16*) carve((size_t)BV * 2048 * 2);

  auto packB = [&](const float* W, bf16* PB, int K, int N) {
    int n = K * N;
    pack_b_kernel<<<(n + 255) / 256, 256, 0, stream>>>(W, PB, K, N);
  };
  packB(Wq,  WqP,  256, 256);
  packB(Wk,  WkP,  256, 256);
  packB(Wv,  WvP,  256, 256);
  packB(Wm,  WmP,  256, 256);
  packB(Wf1, Wf1P, 512, 2048);
  packB(Wf2, Wf2P, 2048, 256);
  {
    int n4 = BV * 256 / 4;
    cvt_f32_bf16_kernel<<<(n4 + 255) / 256, 256, 0, stream>>>(
        (const float4*)source, srcB, n4);
  }

  // q = src @ Wq   (f32 out)
  gemm16_bf16_kernel<<<dim3(BV / 32, 1), 256, 0, stream>>>(
      srcB, srcB, 256, 256, WqP, 256, qF, nullptr, nullptr, nullptr, nullptr, 0);

  // fused k/v projection + attention (2 voxels per workgroup)
  attn_kernel<<<BV / 2, 256, 0, stream>>>(target, qF, wts, confid, WkP, WvP, aoB);

  // message = LN(out @ Wm) -> bf16
  gemm16_bf16_kernel<<<dim3(BV / 32, 1), 256, 0, stream>>>(
      aoB, aoB, 256, 256, WmP, 256, nullptr, m1B, g1, b1, nullptr, 1);

  // h = gelu([src | message] @ Wf1) -> bf16   (N=2048)
  gemm16_bf16_kernel<<<dim3(BV / 32, 8), 256, 0, stream>>>(
      srcB, m1B, 256, 512, Wf1P, 2048, nullptr, hB, nullptr, nullptr, nullptr, 2);

  // out = source + LN(h @ Wf2)   (K=2048)
  gemm16_bf16_kernel<<<dim3(BV / 32, 1), 256, 0, stream>>>(
      hB, hB, 2048, 2048, Wf2P, 256, (float*)d_out, nullptr, g2, b2, source, 3);
}
